// GoSpecificWattiPooling_15040975470636
// MI455X (gfx1250) — matrix operand order repaired and checked by reading, not compiled, not run on hardware
//
#include <hip/hip_runtime.h>
#include <stdint.h>

// Problem constants (from reference)
#define B_   4
#define L_   4096
#define DH_  1024
#define T_   512
#define DG_  768
#define DP_  256
#define SCALE_ 0.0625f   // DP^-0.5 = 1/16

typedef __attribute__((ext_vector_type(16))) __bf16 v16bf;
typedef __attribute__((ext_vector_type(8)))  float  v8f;
typedef __attribute__((ext_vector_type(4)))  uint32_t u32x4;
typedef __attribute__((ext_vector_type(8)))  int      i32x8;
typedef __attribute__((ext_vector_type(4)))  int      i32x4;

union FragB { v16bf v; uint4 q[2]; };

__device__ __forceinline__ uint16_t f32_to_bf16(float f) {
  uint32_t x = __builtin_bit_cast(uint32_t, f);
  x += 0x7FFFu + ((x >> 16) & 1u);            // round-to-nearest-even
  return (uint16_t)(x >> 16);
}

// Packed f32x2 -> bf16x2 (v_cvt_pk_bf16_f32 when available)
__device__ __forceinline__ uint32_t pack_bf16x2(float lo, float hi) {
#if __has_builtin(__builtin_amdgcn_cvt_pk_bf16_f32)
  typedef __attribute__((ext_vector_type(2))) __bf16 v2bf;
  v2bf r = __builtin_amdgcn_cvt_pk_bf16_f32(lo, hi);
  return __builtin_bit_cast(uint32_t, r);
#else
  return (uint32_t)f32_to_bf16(lo) | ((uint32_t)f32_to_bf16(hi) << 16);
#endif
}

__device__ __forceinline__ v8f vzero8() {
  v8f z;
#pragma unroll
  for (int i = 0; i < 8; ++i) z[i] = 0.0f;
  return z;
}

// A-fragment (16x32 bf16, M x K), ISA 7.12.2 layout.
__device__ __forceinline__ v16bf load_a_frag(const uint16_t* base, int stride) {
  const int lane = threadIdx.x & 31;
  const int row  = lane & 15;
  const int ko   = (lane >> 4) << 3;          // 0 or 8
  const uint16_t* p = base + row * stride;
  FragB f;
  f.q[0] = *reinterpret_cast<const uint4*>(p + ko);
  f.q[1] = *reinterpret_cast<const uint4*>(p + 16 + ko);
  return f.v;
}

// B-fragment (32x16 bf16, K x N) from an N-major (transposed) tile.
__device__ __forceinline__ v16bf load_bt_frag(const uint16_t* base, int stride) {
  const int lane = threadIdx.x & 31;
  const int col  = lane & 15;
  const int ko   = (lane >> 4) << 3;          // 0 or 8
  const uint16_t* p = base + col * stride;
  FragB f;
  f.q[0] = *reinterpret_cast<const uint4*>(p + ko);
  f.q[1] = *reinterpret_cast<const uint4*>(p + 16 + ko);
  return f.v;
}

__device__ __forceinline__ v8f wmma_bf16(v16bf a, v16bf b, v8f c) {
  return __builtin_amdgcn_wmma_f32_16x16x32_bf16(false, a, false, b, (short)0, c,
                                                 false, false);
}

__device__ __forceinline__ float red_max16(float v) {
#pragma unroll
  for (int m = 1; m < 16; m <<= 1) v = fmaxf(v, __shfl_xor(v, m, 32));
  return v;
}
__device__ __forceinline__ float red_sum16(float v) {
#pragma unroll
  for (int m = 1; m < 16; m <<= 1) v += __shfl_xor(v, m, 32);
  return v;
}

// Async global->LDS copy of 16B per lane (ASYNCcnt path, gfx1250).
__device__ __forceinline__ void async_copy_b128(uint32_t lds_off, const void* gptr) {
  asm volatile("global_load_async_to_lds_b128 %0, %1, off"
               :: "v"(lds_off), "v"(gptr)
               : "memory");
}
__device__ __forceinline__ void wait_asynccnt0() {
  asm volatile("s_wait_asynccnt 0x0" ::: "memory");
}

// TDM: issue a 2D bf16 tile load (tile_d0 x tile_d1 rows, row stride strideElem)
// from global `gaddr` into LDS offset `lds_off` (packed rows).
__device__ __forceinline__ void tdm_load_tile_2d(uint64_t gaddr, uint32_t lds_off,
                                                 uint32_t tile_d0, uint32_t tile_d1,
                                                 uint32_t stride_elem,
                                                 uint32_t tensor_d0, uint32_t tensor_d1) {
  u32x4 g0;
  g0[0] = 1u;                                   // count=1 (valid user descriptor)
  g0[1] = lds_off;                              // lds_addr (bytes)
  g0[2] = (uint32_t)gaddr;                      // global_addr[31:0]
  g0[3] = (uint32_t)((gaddr >> 32) & 0x1FFFFFFu) | (2u << 30); // addr[56:32], type=2
  uint32_t g1w[8];
  g1w[0] = (1u << 16);                          // workgroup_mask=0, data_size=1 (2B)
  g1w[1] = (tensor_d0 & 0xFFFFu) << 16;         // tensor_dim0[15:0]
  g1w[2] = (tensor_d0 >> 16) | ((tensor_d1 & 0xFFFFu) << 16);
  g1w[3] = (tensor_d1 >> 16) | (tile_d0 << 16); // tile_dim0
  g1w[4] = tile_d1;                             // tile_dim1 (tile_dim2=0)
  g1w[5] = stride_elem;                         // tensor_dim0_stride[31:0]
  g1w[6] = 0u;                                  // stride[47:32], dim1_stride[15:0]
  g1w[7] = 0u;
  i32x8 g1;
#pragma unroll
  for (int i = 0; i < 8; ++i) g1[i] = (int)g1w[i];
#if __has_builtin(__builtin_amdgcn_tensor_load_to_lds)
  i32x4 gz4; gz4[0] = 0; gz4[1] = 0; gz4[2] = 0; gz4[3] = 0;
  i32x8 gz8;
#pragma unroll
  for (int i = 0; i < 8; ++i) gz8[i] = 0;
  // 6-arg variant (therock/clang-23 headers): (g0, g1, g2, g3, g4, cpol)
  __builtin_amdgcn_tensor_load_to_lds(g0, g1, gz4, gz4, gz8, 0);
#else
  asm volatile("tensor_load_to_lds %0, %1" :: "s"(g0), "s"(g1) : "memory");
#endif
}

// ---------------------------------------------------------------------------
// Kernel 1: f32 -> bf16 elementwise (H copy used as V in the P*H GEMM)
// ---------------------------------------------------------------------------
__global__ __launch_bounds__(256) void k_cvt_bf16(const float* __restrict__ in,
                                                  uint32_t* __restrict__ out,
                                                  int n4) {
  int i = blockIdx.x * 256 + threadIdx.x;
  if (i < n4) {
    float4 f = reinterpret_cast<const float4*>(in)[i];
    uint2 o;
    o.x = pack_bf16x2(f.x, f.y);
    o.y = pack_bf16x2(f.z, f.w);
    reinterpret_cast<uint2*>(out)[i] = o;
  }
}

// ---------------------------------------------------------------------------
// Kernel 2: projection GEMM  C[M,N](bf16) = A[M,Ka](f32) * B[Ka,N](f32)
// Block tile 128x64, 8 waves (4x2), wave tile 32x32 (2x2 WMMA frags), K-step 32.
// ---------------------------------------------------------------------------
__global__ __launch_bounds__(256) void k_proj_gemm(const float* __restrict__ A,
                                                   const float* __restrict__ Bm,
                                                   uint16_t* __restrict__ C,
                                                   int M, int Ka, int N) {
  __shared__ uint16_t As[128][32];   // row-major (M x K)
  __shared__ uint16_t Bt[64][32];    // N-major   (N x K)
  const int gm0 = blockIdx.y * 128;
  const int gn0 = blockIdx.x * 64;
  const int tid = threadIdx.x;
  const int w = tid >> 5, lane = tid & 31;
  const int mw = w >> 1, nw = w & 1;
  (void)M;

  v8f acc[2][2];
#pragma unroll
  for (int i = 0; i < 2; ++i)
#pragma unroll
    for (int j = 0; j < 2; ++j) acc[i][j] = vzero8();

  for (int k0 = 0; k0 < Ka; k0 += 32) {
#pragma unroll
    for (int it = 0; it < 4; ++it) {            // stage A: 128x32 f32 -> bf16
      int idx = tid + 256 * it;
      int row = idx >> 3, ch = idx & 7;
      float4 f = *reinterpret_cast<const float4*>(
          A + (size_t)(gm0 + row) * Ka + k0 + ch * 4);
      uint2 h2;
      h2.x = pack_bf16x2(f.x, f.y);
      h2.y = pack_bf16x2(f.z, f.w);
      *reinterpret_cast<uint2*>(&As[row][ch * 4]) = h2;
    }
#pragma unroll
    for (int it = 0; it < 2; ++it) {            // stage B transposed
      int idx = tid + 256 * it;
      int k = idx >> 4, nc = idx & 15;
      float4 f = *reinterpret_cast<const float4*>(
          Bm + (size_t)(k0 + k) * N + gn0 + nc * 4);
      Bt[nc * 4 + 0][k] = f32_to_bf16(f.x);
      Bt[nc * 4 + 1][k] = f32_to_bf16(f.y);
      Bt[nc * 4 + 2][k] = f32_to_bf16(f.z);
      Bt[nc * 4 + 3][k] = f32_to_bf16(f.w);
    }
    __syncthreads();
    v16bf af[2], bfg[2];
#pragma unroll
    for (int i = 0; i < 2; ++i) af[i]  = load_a_frag(&As[mw * 32 + i * 16][0], 32);
#pragma unroll
    for (int j = 0; j < 2; ++j) bfg[j] = load_bt_frag(&Bt[nw * 32 + j * 16][0], 32);
#pragma unroll
    for (int i = 0; i < 2; ++i)
#pragma unroll
      for (int j = 0; j < 2; ++j) acc[i][j] = wmma_bf16(af[i], bfg[j], acc[i][j]);
    __syncthreads();
  }

  const int crow = (lane >> 4) << 3, ccol = lane & 15;
#pragma unroll
  for (int i = 0; i < 2; ++i)
#pragma unroll
    for (int j = 0; j < 2; ++j)
#pragma unroll
      for (int r = 0; r < 8; ++r) {
        int row = gm0 + mw * 32 + i * 16 + crow + r;
        int col = gn0 + nw * 32 + j * 16 + ccol;
        C[(size_t)row * N + col] = f32_to_bf16(acc[i][j][r]);
      }
}

// ---------------------------------------------------------------------------
// Kernels 3/4 share: per-wave async staging of a 16x256 bf16 K-tile into LDS.
// Each wave issues 16 b128 async loads (16 rows x 512B), waits ASYNCcnt.
// ---------------------------------------------------------------------------
__device__ __forceinline__ void stage_k_tile_async(uint16_t* ldsbuf,
                                                   const uint16_t* kb, int l0,
                                                   int lane) {
  const uint32_t lds_base = (uint32_t)(uintptr_t)ldsbuf;
  const char* gbase = (const char*)(kb + (size_t)l0 * DP_);
#pragma unroll
  for (int it = 0; it < 16; ++it) {
    uint32_t off = (uint32_t)(it * 512 + lane * 16);
    async_copy_b128(lds_base + off, gbase + off);
  }
  wait_asynccnt0();
}

// ---------------------------------------------------------------------------
// Kernel 3: softmax statistics (online max/sumexp per row of Q*K^T).
// ---------------------------------------------------------------------------
__global__ __launch_bounds__(256) void k_attn_stats(const uint16_t* __restrict__ Q,
                                                    const uint16_t* __restrict__ K,
                                                    const uint8_t* __restrict__ mask,
                                                    float* __restrict__ stats) {
  __shared__ uint16_t Kt[8][16 * DP_];   // 64 KiB, one buffer per wave
  const int tile = blockIdx.x;
  const int b  = tile >> 5;              // 32 t-tiles per batch
  const int t0 = (tile & 31) << 4;
  const int w = threadIdx.x >> 5, lane = threadIdx.x & 31;
  const int crow = (lane >> 4) << 3, ccol = lane & 15;

  v16bf qa[8];
  const uint16_t* qbase = Q + (size_t)(b * T_ + t0) * DP_;
#pragma unroll
  for (int pc = 0; pc < 8; ++pc) qa[pc] = load_a_frag(qbase + pc * 32, DP_);

  float m[8], s[8];
#pragma unroll
  for (int r = 0; r < 8; ++r) { m[r] = -INFINITY; s[r] = 0.0f; }

  const uint16_t* kb = K + (size_t)b * L_ * DP_;
  const uint8_t*  mb = mask + (size_t)b * L_;

  for (int lt = w; lt < L_ / 16; lt += 8) {
    const int l0 = lt << 4;
    stage_k_tile_async(&Kt[w][0], kb, l0, lane);
    v8f acc = vzero8();
#pragma unroll
    for (int pc = 0; pc < 8; ++pc) {
      v16bf bfg = load_bt_frag(&Kt[w][pc * 32], DP_);
      acc = wmma_bf16(qa[pc], bfg, acc);
    }
    const bool msk = mb[l0 + ccol] != 0;
#pragma unroll
    for (int r = 0; r < 8; ++r) {
      float v = msk ? -INFINITY : acc[r] * SCALE_;
      float rowmax = red_max16(v);
      float m_new  = fmaxf(m[r], rowmax);
      float e      = __expf(v - m_new);
      float rowsum = red_sum16(e);
      s[r] = s[r] * __expf(m[r] - m_new) + rowsum;
      m[r] = m_new;
    }
  }

  __shared__ float rm[8][16], rs[8][16];
  if ((lane & 15) == 0) {
#pragma unroll
    for (int r = 0; r < 8; ++r) { rm[w][crow + r] = m[r]; rs[w][crow + r] = s[r]; }
  }
  __syncthreads();
  if (threadIdx.x < 16) {
    int row = threadIdx.x;
    float M = -INFINITY;
#pragma unroll
    for (int ww = 0; ww < 8; ++ww) M = fmaxf(M, rm[ww][row]);
    float S = 0.0f;
#pragma unroll
    for (int ww = 0; ww < 8; ++ww) S += rs[ww][row] * __expf(rm[ww][row] - M);
    stats[(size_t)(b * T_ + t0 + row) * 2 + 0] = M;
    stats[(size_t)(b * T_ + t0 + row) * 2 + 1] = S;
  }
}

// ---------------------------------------------------------------------------
// Kernel 4: P = softmax weights (bf16), recomputing logits and normalizing.
// ---------------------------------------------------------------------------
__global__ __launch_bounds__(256) void k_attn_p(const uint16_t* __restrict__ Q,
                                                const uint16_t* __restrict__ K,
                                                const uint8_t* __restrict__ mask,
                                                const float* __restrict__ stats,
                                                uint16_t* __restrict__ P) {
  __shared__ uint16_t Kt[8][16 * DP_];   // 64 KiB, one buffer per wave
  const int tile = blockIdx.x;
  const int b  = tile >> 5;
  const int t0 = (tile & 31) << 4;
  const int w = threadIdx.x >> 5, lane = threadIdx.x & 31;
  const int crow = (lane >> 4) << 3, ccol = lane & 15;

  v16bf qa[8];
  const uint16_t* qbase = Q + (size_t)(b * T_ + t0) * DP_;
#pragma unroll
  for (int pc = 0; pc < 8; ++pc) qa[pc] = load_a_frag(qbase + pc * 32, DP_);

  float mrow[8], sinv[8];
#pragma unroll
  for (int r = 0; r < 8; ++r) {
    mrow[r] = stats[(size_t)(b * T_ + t0 + crow + r) * 2 + 0];
    sinv[r] = 1.0f / stats[(size_t)(b * T_ + t0 + crow + r) * 2 + 1];
  }

  const uint16_t* kb = K + (size_t)b * L_ * DP_;
  const uint8_t*  mb = mask + (size_t)b * L_;

  for (int lt = w; lt < L_ / 16; lt += 8) {
    const int l0 = lt << 4;
    stage_k_tile_async(&Kt[w][0], kb, l0, lane);
    v8f acc = vzero8();
#pragma unroll
    for (int pc = 0; pc < 8; ++pc) {
      v16bf bfg = load_bt_frag(&Kt[w][pc * 32], DP_);
      acc = wmma_bf16(qa[pc], bfg, acc);
    }
    const bool msk = mb[l0 + ccol] != 0;
#pragma unroll
    for (int r = 0; r < 8; ++r) {
      float v = acc[r] * SCALE_;
      float p = msk ? 0.0f : __expf(v - mrow[r]) * sinv[r];
      P[(size_t)(b * T_ + t0 + crow + r) * L_ + l0 + ccol] = f32_to_bf16(p);
    }
  }
}

// ---------------------------------------------------------------------------
// Kernel 5: Z[B*T, DH](f32) = P[B*T, L](bf16) * Hbf[b][L, DH](bf16)
// A tile (128x32 of P) staged by the Tensor Data Mover; B tile transposed
// by threads. Block tile 128x64, K-step 32 over L.
// ---------------------------------------------------------------------------
__global__ __launch_bounds__(256) void k_z_gemm(const uint16_t* __restrict__ P,
                                                const uint16_t* __restrict__ Hbf,
                                                float* __restrict__ Z) {
  __shared__ uint16_t As[128][32];
  __shared__ uint16_t Bt[64][32];
  const int gm0 = blockIdx.y * 128;       // global row in [0, B*T)
  const int gn0 = blockIdx.x * 64;        // column in [0, DH)
  const int b = gm0 / T_;
  const int tid = threadIdx.x;
  const int w = tid >> 5, lane = tid & 31;
  const int mw = w >> 1, nw = w & 1;

  const uint16_t* Hb = Hbf + (size_t)b * L_ * DH_;

  v8f acc[2][2];
#pragma unroll
  for (int i = 0; i < 2; ++i)
#pragma unroll
    for (int j = 0; j < 2; ++j) acc[i][j] = vzero8();

  for (int k0 = 0; k0 < L_; k0 += 32) {
    if (w == 0) {   // TDM: DMA the 128x32 bf16 A tile of P into LDS
      uint64_t ga = (uint64_t)(uintptr_t)(P + (size_t)gm0 * L_ + k0);
      tdm_load_tile_2d(ga, (uint32_t)(uintptr_t)&As[0][0],
                       /*tile_d0=*/32, /*tile_d1=*/128,
                       /*stride=*/L_,
                       /*tensor_d0=*/L_ - k0, /*tensor_d1=*/B_ * T_);
      __builtin_amdgcn_s_wait_tensorcnt(0);
    }
    {                                     // stage B transposed: 32x64 -> 64x32
      int k = tid >> 3, nc = tid & 7;
      uint4 vv = *reinterpret_cast<const uint4*>(
          Hb + (size_t)(k0 + k) * DH_ + gn0 + nc * 8);
      const uint16_t* e = reinterpret_cast<const uint16_t*>(&vv);
#pragma unroll
      for (int j = 0; j < 8; ++j) Bt[nc * 8 + j][k] = e[j];
    }
    __syncthreads();
    v16bf af[2], bfg[2];
#pragma unroll
    for (int i = 0; i < 2; ++i) af[i]  = load_a_frag(&As[mw * 32 + i * 16][0], 32);
#pragma unroll
    for (int j = 0; j < 2; ++j) bfg[j] = load_bt_frag(&Bt[nw * 32 + j * 16][0], 32);
#pragma unroll
    for (int i = 0; i < 2; ++i)
#pragma unroll
      for (int j = 0; j < 2; ++j) acc[i][j] = wmma_bf16(af[i], bfg[j], acc[i][j]);
    __syncthreads();
  }

  const int crow = (lane >> 4) << 3, ccol = lane & 15;
#pragma unroll
  for (int i = 0; i < 2; ++i)
#pragma unroll
    for (int j = 0; j < 2; ++j)
#pragma unroll
      for (int r = 0; r < 8; ++r) {
        int row = gm0 + mw * 32 + i * 16 + crow + r;
        int col = gn0 + nw * 32 + j * 16 + ccol;
        Z[(size_t)row * DH_ + col] = acc[i][j][r];
      }
}

// ---------------------------------------------------------------------------
extern "C" void kernel_launch(void* const* d_in, const int* in_sizes, int n_in,
                              void* d_out, int out_size, void* d_ws, size_t ws_size,
                              hipStream_t stream) {
  (void)in_sizes; (void)n_in; (void)out_size; (void)ws_size;
  const float*   H    = (const float*)d_in[0];    // [B,L,DH] f32
  const float*   G    = (const float*)d_in[1];    // [B,T,DG] f32
  const uint8_t* mask = (const uint8_t*)d_in[2];  // [B,L] bool
  const float*   Wk   = (const float*)d_in[3];    // [DH,DP] f32
  const float*   Wq   = (const float*)d_in[4];    // [DG,DP] f32
  float* Z = (float*)d_out;                       // [B,T,DH] f32

  char* ws = (char*)d_ws;
  uint16_t* Hbf   = (uint16_t*)(ws);                    // 32 MiB
  uint16_t* Kbf   = (uint16_t*)(ws + 33554432);         //  8 MiB
  uint16_t* Qbf   = (uint16_t*)(ws + 41943040);         //  1 MiB
  uint16_t* Pmat  = (uint16_t*)(ws + 42991616);         // 16 MiB
  float*    stats = (float*)   (ws + 59768832);         // 16 KiB

  // 1) H -> bf16
  const int nH4 = (B_ * L_ * DH_) / 4;                  // 4,194,304
  k_cvt_bf16<<<nH4 / 256, 256, 0, stream>>>(H, (uint32_t*)Hbf, nH4);

  // 2) K = H @ Wk   (M=B*L=16384, Ka=DH=1024, N=DP=256)
  k_proj_gemm<<<dim3(DP_ / 64, (B_ * L_) / 128), 256, 0, stream>>>(
      H, Wk, Kbf, B_ * L_, DH_, DP_);

  // 3) Q = G @ Wq   (M=B*T=2048, Ka=DG=768, N=DP=256)
  k_proj_gemm<<<dim3(DP_ / 64, (B_ * T_) / 128), 256, 0, stream>>>(
      G, Wq, Qbf, B_ * T_, DG_, DP_);

  // 4) softmax stats, then normalized P (bf16)
  k_attn_stats<<<B_ * (T_ / 16), 256, 0, stream>>>(Qbf, Kbf, mask, stats);
  k_attn_p<<<B_ * (T_ / 16), 256, 0, stream>>>(Qbf, Kbf, mask, stats, Pmat);

  // 5) Z = P @ Hbf
  k_z_gemm<<<dim3(DH_ / 64, (B_ * T_) / 128), 256, 0, stream>>>(Pmat, Hbf, Z);
}
